// TreeModelMDIN_65429531787914
// MI455X (gfx1250) — compile-verified, matrix operand order checked by reading, not run on hardware
//
#include <hip/hip_runtime.h>
#include <hip/hip_bf16.h>
#include <math.h>

// Problem constants from the reference
#define B_  32
#define T_  96
#define H_  512
#define E_  512
#define V_  10000
#define F_  2048
#define XC  2560   // X columns: [x_a | x_f | ha_h | hf_h | att], 5*512
#define GC  2560   // gate columns: [ingate | outgate | cellgate | f1 | f2]

typedef __attribute__((ext_vector_type(16))) __bf16 v16bf;
typedef __attribute__((ext_vector_type(8)))  __bf16 v8bf;
typedef __attribute__((ext_vector_type(8)))  float  v8f;

// ---------------------------------------------------------------------------
// WMMA fragment helpers (wave32, v_wmma_f32_16x16x32_bf16)
//
// 16-bit A-matrix 16x32 layout (ISA 7.12.2): lane L<16 holds M=L with
// K in {0..7} U {16..23}; lane L>=16 holds M=L-16 with K in {8..15} U {24..31}.
// Element e of the v16bf maps to k = (e>>3)*16 + ((L>>4)<<3) + (e&7).
// B (32x16, K-major per output column n=L&15) uses the mirrored layout; we
// pre-pack weights so a B fragment is 16 contiguous bf16 per lane (32B load).
// ---------------------------------------------------------------------------
__device__ __forceinline__ v16bf load_a_frag(const __bf16* A, int row0, int ldk,
                                             int kbase, int lane) {
  const __bf16* p = A + (size_t)(row0 + (lane & 15)) * ldk + kbase + ((lane >> 4) << 3);
  v8bf lo = *reinterpret_cast<const v8bf*>(p);        // k = kbase+half*8 .. +7
  v8bf hi = *reinterpret_cast<const v8bf*>(p + 16);   // k = +16 .. +23
  v16bf r;
#pragma unroll
  for (int j = 0; j < 8; ++j) { r[j] = lo[j]; r[8 + j] = hi[j]; }
  return r;
}

__device__ __forceinline__ float sigmoidf_(float x) { return 1.0f / (1.0f + __expf(-x)); }

// ---------------------------------------------------------------------------
// Pack a row-major f32 weight W[N][K] (out = in . W^T) into bf16 WMMA-B tiles:
// tile (nt,kt) occupies 512 contiguous bf16; lane L owns elements [L*16, L*16+16)
// with n = nt*16 + (L&15), k = kt*32 + (e>>3)*16 + ((L>>4)<<3) + (e&7).
// ---------------------------------------------------------------------------
__global__ void pack_b_kernel(const float* __restrict__ W, __bf16* __restrict__ Bp,
                              int N, int K) {
  const int KT = K >> 5;
  const size_t total = (size_t)N * K;
  for (size_t idx = (size_t)blockIdx.x * blockDim.x + threadIdx.x; idx < total;
       idx += (size_t)gridDim.x * blockDim.x) {
    size_t tile = idx >> 9;
    int within = (int)(idx & 511);
    int lane = within >> 4;
    int e    = within & 15;
    int nt = (int)(tile / KT);
    int kt = (int)(tile - (size_t)nt * KT);
    int n = nt * 16 + (lane & 15);
    int k = kt * 32 + ((e >> 3) << 4) + ((lane >> 4) << 3) + (e & 7);
    Bp[idx] = (__bf16)W[(size_t)n * K + k];
  }
}

__global__ void cvt_f32_bf16(const float* __restrict__ src, __bf16* __restrict__ dst, int n) {
  for (int i = blockIdx.x * blockDim.x + threadIdx.x; i < n; i += gridDim.x * blockDim.x)
    dst[i] = (__bf16)src[i];
}

// ---------------------------------------------------------------------------
// Generic bf16 WMMA GEMM, register-blocked along N:
//   C[M][N] = A[M][K](bf16,row-major) . Bp + bias
// Each wave computes NW adjacent 16x16 N-tiles of one M-tile, sharing the A
// fragment (NW independent wmma chains -> ILP, A traffic / NW).
// Requires (N/16) % NW == 0.
// ---------------------------------------------------------------------------
template <int NW>
__global__ void gemm_bf16_kernel(const __bf16* __restrict__ A, const __bf16* __restrict__ Bp,
                                 const float* __restrict__ bias, float* __restrict__ C,
                                 int M, int N, int K, int ldc) {
  const int lane = threadIdx.x & 31;
  const int wv   = threadIdx.x >> 5;
  const int MT = M >> 4, NT = N >> 4, KT = K >> 5;
  const int NG = NT / NW;
  const int nWaves = (blockDim.x >> 5) * gridDim.x;
  for (int t = blockIdx.x * (blockDim.x >> 5) + wv; t < MT * NG; t += nWaves) {
    const int mt = t / NG, g = t - mt * NG;
    const int nt0 = g * NW;
    const __bf16* bt[NW];
#pragma unroll
    for (int j = 0; j < NW; ++j)
      bt[j] = Bp + ((size_t)(nt0 + j) * KT) * 512 + lane * 16;
    v8f acc[NW] = {};
    for (int kt = 0; kt < KT; ++kt) {
      v16bf a = load_a_frag(A, mt * 16, K, kt * 32, lane);
#pragma unroll
      for (int j = 0; j < NW; ++j) {
        if (kt + 1 < KT) __builtin_prefetch(bt[j] + (size_t)(kt + 1) * 512, 0, 3);
        v16bf b = *reinterpret_cast<const v16bf*>(bt[j] + (size_t)kt * 512);
        acc[j] = __builtin_amdgcn_wmma_f32_16x16x32_bf16(false, a, false, b, (short)0,
                                                         acc[j], false, false);
      }
    }
    // D layout: VGPR r, lanes 0-15 -> M=r, lanes 16-31 -> M=8+r; N = lane&15.
    const int mrow = mt * 16 + ((lane >> 4) << 3);
#pragma unroll
    for (int j = 0; j < NW; ++j) {
      const int n = (nt0 + j) * 16 + (lane & 15);
      const float bv = bias ? bias[n] : 0.0f;
#pragma unroll
      for (int r = 0; r < 8; ++r)
        C[(size_t)(mrow + r) * ldc + n] = acc[j][r] + bv;
    }
  }
}

// ---------------------------------------------------------------------------
// Step 1: gather the concatenated bf16 activation matrix X[32][2560]
// ---------------------------------------------------------------------------
__global__ void step_gather(const int* __restrict__ word_idx, const int* __restrict__ father_idx,
                            const float* __restrict__ embed, const float* __restrict__ h_states,
                            const float* __restrict__ att, __bf16* __restrict__ X, int i) {
  const int total = B_ * XC;
  for (int idx = blockIdx.x * blockDim.x + threadIdx.x; idx < total;
       idx += gridDim.x * blockDim.x) {
    const int b = idx / XC, col = idx - b * XC;
    const int f  = father_idx[b * T_ + i];
    const int ip = (i > 0) ? (i - 1) : 0;
    const bool first = (i == 0);
    const bool sib0  = first || (((i - 1) % 3) == 0);
    const int seg = col >> 9, off = col & 511;
    float v;
    if (seg == 0) {          // x_a = embed[word at father]
      const int wa = word_idx[b * T_ + f];
      v = first ? 0.0f : embed[(size_t)wa * E_ + off];
    } else if (seg == 1) {   // x_f = embed[word at left sibling]
      const int wf = word_idx[b * T_ + ip];
      v = sib0 ? 0.0f : embed[(size_t)wf * E_ + off];
    } else if (seg == 2) {   // ha_h
      v = first ? 0.0f : h_states[((size_t)b * T_ + f) * H_ + off];
    } else if (seg == 3) {   // hf_h
      v = sib0 ? 0.0f : h_states[((size_t)b * T_ + ip) * H_ + off];
    } else {                 // att
      v = att[b * H_ + off];
    }
    X[idx] = (__bf16)v;
  }
}

// ---------------------------------------------------------------------------
// Step 2: all gate GEMMs in one WMMA kernel, 4 N-tiles per wave, fused
// bias + activation.  Gate column map:
//   [0,1024)    = sigmoid(Hc.W_H^T + b_H)        (ingate | outgate), K over X[0,2048)
//   [1024,1536) = tanh([Hc,att].W_I^T + b_I),    K over X[0,2560)
//   [1536,2048) = sigmoid([x_a,ha_h].W_f1^T+b1), K over X[0,512) U X[1024,1536)
//   [2048,2560) = sigmoid([x_f,hf_h].W_f2^T+b2), K over X[512,1024) U X[1536,2048)
// The X-column remap is branch-free: xc = xBase + (kt>>4)*xStride + (kt&15)*32.
// ---------------------------------------------------------------------------
__global__ void step_gates(const __bf16* __restrict__ X,
                           const __bf16* __restrict__ WHp, const __bf16* __restrict__ WIp,
                           const __bf16* __restrict__ Wf1p, const __bf16* __restrict__ Wf2p,
                           const float* __restrict__ bH, const float* __restrict__ bI,
                           const float* __restrict__ bf1, const float* __restrict__ bf2,
                           float* __restrict__ G) {
  const int lane = threadIdx.x & 31;
  const int wv   = threadIdx.x >> 5;
  const int NGROUP = 40;              // (64+32+32+32)/4 tile-groups per M-tile
  const int nTasks = 2 * NGROUP;      // MT = 2 (32 rows)
  const int nWaves = (blockDim.x >> 5) * gridDim.x;
  for (int t = blockIdx.x * (blockDim.x >> 5) + wv; t < nTasks; t += nWaves) {
    const int mt = t / NGROUP, gg = t - mt * NGROUP;
    const __bf16* Bp; const float* bseg;
    int KT, ntL, ocBase, segBase, xBase, xStride;
    bool isTanh = false;
    if (gg < 16) {
      Bp = WHp;  KT = 64; ntL = gg * 4;        ocBase = ntL * 16;
      segBase = 0;    xBase = 0;   xStride = 512;  bseg = bH;
    } else if (gg < 24) {
      Bp = WIp;  KT = 80; ntL = (gg - 16) * 4; ocBase = 1024 + ntL * 16;
      segBase = 1024; xBase = 0;   xStride = 512;  bseg = bI;  isTanh = true;
    } else if (gg < 32) {
      Bp = Wf1p; KT = 32; ntL = (gg - 24) * 4; ocBase = 1536 + ntL * 16;
      segBase = 1536; xBase = 0;   xStride = 1024; bseg = bf1;
    } else {
      Bp = Wf2p; KT = 32; ntL = (gg - 32) * 4; ocBase = 2048 + ntL * 16;
      segBase = 2048; xBase = 512; xStride = 1024; bseg = bf2;
    }
    const __bf16* bt[4];
#pragma unroll
    for (int j = 0; j < 4; ++j)
      bt[j] = Bp + ((size_t)(ntL + j) * KT) * 512 + lane * 16;
    v8f acc[4] = {};
    for (int kt = 0; kt < KT; ++kt) {
      const int xc = xBase + (kt >> 4) * xStride + ((kt & 15) << 5);
      v16bf a = load_a_frag(X, mt * 16, XC, xc, lane);
#pragma unroll
      for (int j = 0; j < 4; ++j) {
        if (kt + 1 < KT) __builtin_prefetch(bt[j] + (size_t)(kt + 1) * 512, 0, 3);
        v16bf b = *reinterpret_cast<const v16bf*>(bt[j] + (size_t)kt * 512);
        acc[j] = __builtin_amdgcn_wmma_f32_16x16x32_bf16(false, a, false, b, (short)0,
                                                         acc[j], false, false);
      }
    }
    const int mrow = mt * 16 + ((lane >> 4) << 3);
#pragma unroll
    for (int j = 0; j < 4; ++j) {
      const int n = ocBase + j * 16 + (lane & 15);
      const float bv = bseg[n - segBase];
#pragma unroll
      for (int r = 0; r < 8; ++r) {
        float v = acc[j][r] + bv;
        v = isTanh ? tanhf(v) : sigmoidf_(v);
        G[(size_t)(mrow + r) * GC + n] = v;
      }
    }
  }
}

// ---------------------------------------------------------------------------
// Step 3: elementwise cell update; store h/c states and bf16 h for logits
// ---------------------------------------------------------------------------
__global__ void step_combine(const float* __restrict__ G, const int* __restrict__ father_idx,
                             float* __restrict__ h_states, float* __restrict__ c_states,
                             __bf16* __restrict__ outBf, int i) {
  const int total = B_ * H_;
  for (int idx = blockIdx.x * blockDim.x + threadIdx.x; idx < total;
       idx += gridDim.x * blockDim.x) {
    const int b = idx / H_, h = idx - b * H_;
    const int f  = father_idx[b * T_ + i];
    const int ip = (i > 0) ? (i - 1) : 0;
    const bool first = (i == 0);
    const bool sib0  = first || (((i - 1) % 3) == 0);
    const float* g = G + (size_t)b * GC;
    const float in_  = g[h];
    const float out_ = g[512 + h];
    const float cell = g[1024 + h];
    const float f1   = g[1536 + h];
    const float f2   = g[2048 + h];
    const float hac = first ? 0.0f : c_states[((size_t)b * T_ + f) * H_ + h];
    const float hfc = sib0 ? 0.0f : c_states[((size_t)b * T_ + ip) * H_ + h];
    const float c  = f1 * hac + f2 * hfc + in_ * cell;
    const float hh = out_ * tanhf(c);
    c_states[((size_t)b * T_ + i) * H_ + h] = c;
    h_states[((size_t)b * T_ + i) * H_ + h] = hh;
    outBf[((size_t)b * T_ + i) * H_ + h] = (__bf16)hh;
  }
}

// ---------------------------------------------------------------------------
// In-place log_softmax over rows of V_ (one block per row)
// ---------------------------------------------------------------------------
__global__ void logsoftmax_kernel(float* __restrict__ out) {
  const int row = blockIdx.x;
  float* p = out + (size_t)row * V_;
  __shared__ float redM[8];
  __shared__ float redS[8];
  __shared__ float bc[2];
  const int lane = threadIdx.x & 31, wv = threadIdx.x >> 5;

  float m = -3.402823466e38f;
  for (int c = threadIdx.x; c < V_; c += blockDim.x) m = fmaxf(m, p[c]);
#pragma unroll
  for (int o = 16; o; o >>= 1) m = fmaxf(m, __shfl_xor(m, o, 32));
  if (lane == 0) redM[wv] = m;
  __syncthreads();
  if (threadIdx.x == 0) {
    float mm = redM[0];
    for (int j = 1; j < 8; ++j) mm = fmaxf(mm, redM[j]);
    bc[0] = mm;
  }
  __syncthreads();
  m = bc[0];

  float s = 0.0f;
  for (int c = threadIdx.x; c < V_; c += blockDim.x) s += __expf(p[c] - m);
#pragma unroll
  for (int o = 16; o; o >>= 1) s += __shfl_xor(s, o, 32);
  if (lane == 0) redS[wv] = s;
  __syncthreads();
  if (threadIdx.x == 0) {
    float ss = 0.0f;
    for (int j = 0; j < 8; ++j) ss += redS[j];
    bc[1] = m + __logf(ss);
  }
  __syncthreads();
  const float lse = bc[1];
  for (int c = threadIdx.x; c < V_; c += blockDim.x) p[c] -= lse;
}

// ---------------------------------------------------------------------------
extern "C" void kernel_launch(void* const* d_in, const int* in_sizes, int n_in,
                              void* d_out, int out_size, void* d_ws, size_t ws_size,
                              hipStream_t stream) {
  const int*   word_idx   = (const int*)d_in[0];
  const int*   father_idx = (const int*)d_in[1];
  const float* fc_feats   = (const float*)d_in[2];
  const float* embed      = (const float*)d_in[3];
  const float* fc_W       = (const float*)d_in[4];
  const float* fc_b       = (const float*)d_in[5];
  const float* W_f1       = (const float*)d_in[6];
  const float* b_f1       = (const float*)d_in[7];
  const float* W_f2       = (const float*)d_in[8];
  const float* b_f2       = (const float*)d_in[9];
  const float* W_H        = (const float*)d_in[10];
  const float* b_H        = (const float*)d_in[11];
  const float* W_I        = (const float*)d_in[12];
  const float* b_I        = (const float*)d_in[13];
  const float* logit_W    = (const float*)d_in[14];
  const float* logit_b    = (const float*)d_in[15];
  float* out = (float*)d_out;

  char* ws = (char*)d_ws;
  size_t off = 0;
  auto alloc = [&](size_t bytes) -> char* {
    char* p = ws + off;
    off = (off + bytes + 255) & ~(size_t)255;
    return p;
  };

  __bf16* WHp       = (__bf16*)alloc((size_t)1024 * 2048 * 2);
  __bf16* WIp       = (__bf16*)alloc((size_t)512 * 2560 * 2);
  __bf16* Wf1p      = (__bf16*)alloc((size_t)512 * 1024 * 2);
  __bf16* Wf2p      = (__bf16*)alloc((size_t)512 * 1024 * 2);
  __bf16* logitWp   = (__bf16*)alloc((size_t)V_ * H_ * 2);
  __bf16* fcWp      = (__bf16*)alloc((size_t)E_ * F_ * 2);
  __bf16* fcFeatsBf = (__bf16*)alloc((size_t)B_ * F_ * 2);
  float*  attBuf    = (float*)alloc((size_t)B_ * H_ * 4);
  __bf16* Xbf       = (__bf16*)alloc((size_t)B_ * XC * 2);
  float*  G         = (float*)alloc((size_t)B_ * GC * 4);
  float*  h_states  = (float*)alloc((size_t)B_ * T_ * H_ * 4);
  float*  c_states  = (float*)alloc((size_t)B_ * T_ * H_ * 4);
  __bf16* outBf     = (__bf16*)alloc((size_t)B_ * T_ * H_ * 2);
  (void)ws_size; (void)in_sizes; (void)n_in; (void)out_size;

  // --- one-time weight packing (re-done every call: deterministic) ---
  pack_b_kernel<<<512, 256, 0, stream>>>(W_H, WHp, 1024, 2048);
  pack_b_kernel<<<512, 256, 0, stream>>>(W_I, WIp, 512, 2560);
  pack_b_kernel<<<256, 256, 0, stream>>>(W_f1, Wf1p, 512, 1024);
  pack_b_kernel<<<256, 256, 0, stream>>>(W_f2, Wf2p, 512, 1024);
  pack_b_kernel<<<1024, 256, 0, stream>>>(logit_W, logitWp, V_, H_);
  pack_b_kernel<<<512, 256, 0, stream>>>(fc_W, fcWp, E_, F_);
  cvt_f32_bf16<<<64, 256, 0, stream>>>(fc_feats, fcFeatsBf, B_ * F_);

  // att = fc_feats . fc_W^T + fc_b  (M=32, N=512, K=2048), NT=32 -> NW=4
  gemm_bf16_kernel<4><<<2, 256, 0, stream>>>(fcFeatsBf, fcWp, fc_b, attBuf, B_, H_, F_, H_);

  // --- sequential tree recurrence ---
  for (int i = 0; i < T_; ++i) {
    step_gather<<<160, 256, 0, stream>>>(word_idx, father_idx, embed, h_states, attBuf,
                                         Xbf, i);
    step_gates<<<10, 256, 0, stream>>>(Xbf, WHp, WIp, Wf1p, Wf2p, b_H, b_I, b_f1, b_f2, G);
    step_combine<<<32, 256, 0, stream>>>(G, father_idx, h_states, c_states, outBf, i);
  }

  // logits = outputs . logit_W^T + logit_b  (M=3072, N=10000, K=512), NT=625 -> NW=5
  gemm_bf16_kernel<5><<<512, 256, 0, stream>>>(outBf, logitWp, logit_b, out,
                                               B_ * T_, V_, H_, V_);
  // in-place log_softmax over V
  logsoftmax_kernel<<<B_ * T_, 256, 0, stream>>>(out);
}
